// AC_Filter_PreNorm_Net_ND_37529424233398
// MI455X (gfx1250) — compile-verified, matrix operand order checked by reading, not run on hardware
//
#include <hip/hip_runtime.h>
#include <hip/hip_bf16.h>

typedef __attribute__((ext_vector_type(16))) _Float16 v16h;
typedef __attribute__((ext_vector_type(8)))  float    v8f;
typedef __attribute__((ext_vector_type(2)))  _Float16 h2;

#define B_   16
#define L_   2048
#define E_   64
#define D_   63
#define DT_  0.01f
#define EPS_ 1e-5f

__device__ __forceinline__ v8f wmma16(v16h a, v16h b, v8f c) {
  // D = A(16x32 f16) x B(32x16 f16) + C(16x16 f32)
  return __builtin_amdgcn_wmma_f32_16x16x32_f16(false, a, false, b,
                                                (short)0, c, false, false);
}

// Load a 16x32 f16 fragment (ISA 16-bit A layout; also used for the B operand
// under the mirrored lane==N convention) from a row-major tile, row stride
// `ld` halves. Per lane: row = lane&15; K pairs per the documented packing.
__device__ __forceinline__ v16h frag_ld(const _Float16* base, int ld) {
  const int lane = threadIdx.x & 31;
  const int r    = lane & 15;
  const int hb   = (lane >> 4) << 3;   // +8 K offset for upper half-wave
  v16h f;
#pragma unroll
  for (int j = 0; j < 8; ++j) {
    const int kb = ((j & 4) << 2) + hb + ((j & 3) << 1); // j<4:K0-15, j>=4:K16-31
    h2 p = *(const h2*)(base + r * ld + kb);
    f[2*j] = p[0]; f[2*j+1] = p[1];
  }
  return f;
}

// ---------------------------------------------------------------- k_prep ----
// wh16[512][64] = f16(Wg*Mg) zero-padded; w_in16[192][64]; ow16[64][64].
__global__ __launch_bounds__(256) void k_prep(
    const float* __restrict__ in_w, const float* __restrict__ out_w,
    const float* __restrict__ Wg, const float* __restrict__ Mg,
    _Float16* __restrict__ w_in16, _Float16* __restrict__ ow16,
    _Float16* __restrict__ wh16) {
  const int tid = blockIdx.x * 256 + threadIdx.x;   // 0 .. 512*64-1
  if (tid < 512 * 64) {
    float v = 0.f;
    if (tid < 504 * 64) v = Wg[tid] * Mg[tid];      // [D*W1, E] flat
    wh16[tid] = (_Float16)v;
  }
  if (tid < 192 * 64) w_in16[tid] = (_Float16)in_w[tid];
  if (tid < 64 * 64)  ow16[tid]  = (_Float16)out_w[tid];
}

// ----------------------------------------------------------------- k_qkv ----
// qkv = prenorm(x) @ in_proj_w^T + b ; q pre-scaled by 1/sqrt(E)=0.125.
__global__ __launch_bounds__(128) void k_qkv(
    const float* __restrict__ inputs, const float* __restrict__ in_b,
    const float* __restrict__ sigma, const _Float16* __restrict__ w16,
    _Float16* __restrict__ q16, _Float16* __restrict__ k16,
    _Float16* __restrict__ v16p) {
  const int lane = threadIdx.x & 31, wid = threadIdx.x >> 5;
  const int r = lane & 15, half = lane >> 4;
  const int m0 = blockIdx.x * 64 + wid * 16;

  // A fragments with pre-norm folded in (col 0 untouched, else /(sigma+eps))
  v16h a0, a1;
  const float* xp = inputs + (size_t)(m0 + r) * E_;
#pragma unroll
  for (int j = 0; j < 8; ++j) {
    const int kb = ((j & 4) << 2) + (half << 3) + ((j & 3) << 1);
    float e0 = xp[kb], e1 = xp[kb + 1];
    if (kb > 0) e0 /= (sigma[kb - 1] + EPS_);
    e1 /= (sigma[kb] + EPS_);
    a0[2*j] = (_Float16)e0; a0[2*j+1] = (_Float16)e1;
    const int kc = kb + 32;
    float f0 = xp[kc] / (sigma[kc - 1] + EPS_);
    float f1 = xp[kc + 1] / (sigma[kc] + EPS_);
    a1[2*j] = (_Float16)f0; a1[2*j+1] = (_Float16)f1;
  }

  for (int t = 0; t < 12; ++t) {
    v16h b0 = frag_ld(w16 + t * 16 * E_, E_);
    v16h b1 = frag_ld(w16 + t * 16 * E_ + 32, E_);
    v8f c = {};
    c = wmma16(a0, b0, c);
    c = wmma16(a1, b1, c);
    const int col = t * 16 + r;
    const float bias = in_b[col];
    const float qs = (t < 4) ? 0.125f : 1.0f;
    _Float16* dst = (t < 4) ? q16 : ((t < 8) ? k16 : v16p);
    const int cl = col & 63;
#pragma unroll
    for (int i = 0; i < 8; ++i) {
      const int row = m0 + i + 8 * half;
      dst[(size_t)row * E_ + cl] = (_Float16)((c[i] + bias) * qs);
    }
  }
}

// ---------------------------------------------------------------- k_attn ----
// Causal flash attention: block = 64 query rows (4 waves x 16), key tiles of 32.
__global__ __launch_bounds__(128) void k_attn(
    const _Float16* __restrict__ q16, const _Float16* __restrict__ k16,
    const _Float16* __restrict__ v16p, float* __restrict__ ao) {
  __shared__ __align__(16) _Float16 Kt[32 * 64];      // [key][e]
  __shared__ __align__(16) _Float16 Vt[64 * 32];      // [e][key] (transposed)
  __shared__ __align__(16) _Float16 Pb[4 * 16 * 32];  // per-wave P repack

  const int b = blockIdx.y, q0 = blockIdx.x * 64;
  const int tid = threadIdx.x, lane = tid & 31, wid = tid >> 5;
  const int r = lane & 15, half = lane >> 4;
  const int qrow = q0 + wid * 16;

  const _Float16* qp = q16 + ((size_t)b * L_ + qrow) * E_;
  v16h aq0 = frag_ld(qp, E_);
  v16h aq1 = frag_ld(qp + 32, E_);

  v8f O0 = {}, O1 = {}, O2 = {}, O3 = {};
  float M[8], Ls[8];
#pragma unroll
  for (int i = 0; i < 8; ++i) { M[i] = -1e30f; Ls[i] = 0.f; }

  const unsigned ldsK = (unsigned)(unsigned long long)(void*)Kt;

  const int nk = (q0 + 64) >> 5;
  for (int kt = 0; kt < nk; ++kt) {
    const int kb = kt * 32;
    __syncthreads();
    // Stage K tile with CDNA5 async global->LDS DMA (no VGPR round-trip).
    // 4KB tile = 256 x b128 chunks; 128 threads x 2 chunks.
    {
      const char* gsrc = (const char*)(k16 + ((size_t)b * L_ + kb) * E_);
#pragma unroll
      for (int u = 0; u < 2; ++u) {
        const int byt = (tid + 128 * u) * 16;
        unsigned long long ga = (unsigned long long)(gsrc + byt);
        unsigned la = ldsK + byt;
        asm volatile("global_load_async_to_lds_b128 %0, %1, off"
                     :: "v"(la), "v"(ga) : "memory");
      }
    }
    // Stage V tile transposed ([e][key]) via VGPRs.
    const _Float16* vsrc = v16p + ((size_t)b * L_ + kb) * E_;
#pragma unroll
    for (int u = 0; u < 8; ++u) {
      const int j = tid + 128 * u, row = j >> 5, cp = (j & 31) * 2;
      Vt[cp * 32 + row]       = vsrc[row * E_ + cp];
      Vt[(cp + 1) * 32 + row] = vsrc[row * E_ + cp + 1];
    }
    asm volatile("s_wait_asynccnt 0x0" ::: "memory");
    __syncthreads();

    // Scores: two 16x16 column-halves, each = 2 WMMA k-steps
    v8f S[2];
#pragma unroll
    for (int h = 0; h < 2; ++h) {
      v16h bk0 = frag_ld(Kt + (16 * h) * E_, E_);
      v16h bk1 = frag_ld(Kt + (16 * h) * E_ + 32, E_);
      v8f s = {};
      s = wmma16(aq0, bk0, s);
      s = wmma16(aq1, bk1, s);
      const int col = kb + 16 * h + r;
#pragma unroll
      for (int i = 0; i < 8; ++i)
        if (col > qrow + i + 8 * half) s[i] = -1e30f;   // causal mask
      S[h] = s;
    }

    // Online softmax (row reductions across the 16-lane column groups)
#pragma unroll
    for (int i = 0; i < 8; ++i) {
      float t = fmaxf(S[0][i], S[1][i]);
      t = fmaxf(t, __shfl_xor(t, 1, 32));
      t = fmaxf(t, __shfl_xor(t, 2, 32));
      t = fmaxf(t, __shfl_xor(t, 4, 32));
      t = fmaxf(t, __shfl_xor(t, 8, 32));
      const float mn = fmaxf(M[i], t);
      const float sc = __expf(M[i] - mn);
      M[i] = mn;
      const float p0 = __expf(S[0][i] - mn), p1 = __expf(S[1][i] - mn);
      S[0][i] = p0; S[1][i] = p1;
      float rs = p0 + p1;
      rs += __shfl_xor(rs, 1, 32);
      rs += __shfl_xor(rs, 2, 32);
      rs += __shfl_xor(rs, 4, 32);
      rs += __shfl_xor(rs, 8, 32);
      Ls[i] = Ls[i] * sc + rs;
      O0[i] *= sc; O1[i] *= sc; O2[i] *= sc; O3[i] *= sc;
    }

    // Repack P (D-layout -> A-layout) through wave-private LDS
    _Float16* pw = Pb + wid * 512;
#pragma unroll
    for (int i = 0; i < 8; ++i) {
      const int row = i + 8 * half;
      pw[row * 32 + r]      = (_Float16)S[0][i];
      pw[row * 32 + 16 + r] = (_Float16)S[1][i];
    }
    asm volatile("s_wait_dscnt 0x0" ::: "memory");
    v16h pa = frag_ld(pw, 32);

    O0 = wmma16(pa, frag_ld(Vt + 0 * 32, 32),  O0);
    O1 = wmma16(pa, frag_ld(Vt + 16 * 32, 32), O1);
    O2 = wmma16(pa, frag_ld(Vt + 32 * 32, 32), O2);
    O3 = wmma16(pa, frag_ld(Vt + 48 * 32, 32), O3);
  }

  float* aop = ao + ((size_t)b * L_ + qrow) * E_;
#pragma unroll
  for (int i = 0; i < 8; ++i) {
    const float inv = 1.0f / Ls[i];
    const int row = i + 8 * half;
    aop[row * E_ + r]      = O0[i] * inv;
    aop[row * E_ + 16 + r] = O1[i] * inv;
    aop[row * E_ + 32 + r] = O2[i] * inv;
    aop[row * E_ + 48 + r] = O3[i] * inv;
  }
}

// --------------------------------------------------------------- k_euler ----
// out-proj + 4 Euler steps of the arithmetic-circuit channels.
__global__ __launch_bounds__(128) void k_euler(
    const float* __restrict__ ao, const float* __restrict__ out_b,
    const _Float16* __restrict__ ow16, const _Float16* __restrict__ wh16,
    const float* __restrict__ bg, const float* __restrict__ sigma,
    float* __restrict__ out) {
  __shared__ __align__(16) _Float16 st16[4 * 16 * 64];  // per-wave state f16
  __shared__ __align__(16) float    vbuf[4 * 16 * 64];  // per-wave v_state*DT

  const int lane = threadIdx.x & 31, wid = threadIdx.x >> 5;
  const int r = lane & 15, half = lane >> 4;
  const int m0 = blockIdx.x * 64 + wid * 16;

  // ---- out_proj: ao @ ow^T + b, then pin col 0 to 1.0 -> state registers
  v16h a0, a1;
  const float* ap = ao + (size_t)(m0 + r) * E_;
#pragma unroll
  for (int j = 0; j < 8; ++j) {
    const int kb = ((j & 4) << 2) + (half << 3) + ((j & 3) << 1);
    a0[2*j] = (_Float16)ap[kb];      a0[2*j+1] = (_Float16)ap[kb + 1];
    a1[2*j] = (_Float16)ap[kb + 32]; a1[2*j+1] = (_Float16)ap[kb + 33];
  }
  v8f St[4];
#pragma unroll
  for (int t = 0; t < 4; ++t) {
    v16h b0 = frag_ld(ow16 + t * 16 * E_, E_);
    v16h b1 = frag_ld(ow16 + t * 16 * E_ + 32, E_);
    v8f c = {};
    c = wmma16(a0, b0, c);
    c = wmma16(a1, b1, c);
    const int col = t * 16 + r;
    const float bias = out_b[col];
#pragma unroll
    for (int i = 0; i < 8; ++i) St[t][i] = (col == 0) ? 1.0f : (c[i] + bias);
  }

  _Float16* sw = st16 + wid * 1024;
  float*    vw = vbuf + wid * 1024;

  for (int it = 0; it < 4; ++it) {
    // state -> LDS f16 (D-layout stores), then repack into A fragments
#pragma unroll
    for (int t = 0; t < 4; ++t)
#pragma unroll
      for (int i = 0; i < 8; ++i)
        sw[(i + 8 * half) * 64 + t * 16 + r] = (_Float16)St[t][i];
    asm volatile("s_wait_dscnt 0x0" ::: "memory");
    v16h sa0 = frag_ld(sw, 64);
    v16h sa1 = frag_ld(sw + 32, 64);

    // H = state @ WH^T (+bg), product over groups of 8 columns -> v_state
    for (int t = 0; t < 32; ++t) {
      __builtin_prefetch(wh16 + ((t + 1) & 31) * 16 * E_, 0, 3);
      v16h b0 = frag_ld(wh16 + t * 16 * E_, E_);      // L2-resident weights
      v16h b1 = frag_ld(wh16 + t * 16 * E_ + 32, E_);
      v8f h = {};
      h = wmma16(sa0, b0, h);
      h = wmma16(sa1, b1, h);
      const int c = t * 16 + r, d = c >> 3, w = c & 7;
      const float bgv = (d < D_) ? bg[d * 8 + w] : 0.f;
#pragma unroll
      for (int i = 0; i < 8; ++i) {
        float p = h[i] + bgv;
        p *= __shfl_xor(p, 1, 32);
        p *= __shfl_xor(p, 2, 32);
        p *= __shfl_xor(p, 4, 32);   // product over the 8-lane (w) group
        if ((lane & 7) == 0) {
          const int dd = t * 2 + ((lane >> 3) & 1);
          vw[(i + 8 * half) * 64 + dd] = p * DT_;
        }
      }
    }
    asm volatile("s_wait_dscnt 0x0" ::: "memory");

    // Euler update + denormalized output write
#pragma unroll
    for (int t = 0; t < 4; ++t)
#pragma unroll
      for (int i = 0; i < 8; ++i) {
        const int col = t * 16 + r;
        if (col > 0) {
          const int d = col - 1, row = i + 8 * half;
          const float ns = St[t][i] + vw[row * 64 + d];
          St[t][i] = ns;
          out[(size_t)(m0 + row) * (4 * D_) + it * D_ + d] =
              ns * (sigma[d] + EPS_);
        }
      }
  }
}

extern "C" void kernel_launch(void* const* d_in, const int* in_sizes, int n_in,
                              void* d_out, int out_size, void* d_ws, size_t ws_size,
                              hipStream_t stream) {
  (void)in_sizes; (void)n_in; (void)out_size; (void)ws_size;
  const float* inputs = (const float*)d_in[1];
  const float* in_w   = (const float*)d_in[2];
  const float* in_b   = (const float*)d_in[3];
  const float* out_w  = (const float*)d_in[4];
  const float* out_b  = (const float*)d_in[5];
  const float* Wg     = (const float*)d_in[6];
  const float* Mg     = (const float*)d_in[7];
  const float* bg     = (const float*)d_in[8];
  const float* sigma  = (const float*)d_in[9];
  float* out = (float*)d_out;

  // Workspace layout (bytes, 256-aligned): all intermediates fit in 192MB L2.
  char* w = (char*)d_ws;
  _Float16* w_in16 = (_Float16*)(w + 0);         //  24576 B
  _Float16* ow16   = (_Float16*)(w + 24576);     //   8192 B
  _Float16* wh16   = (_Float16*)(w + 32768);     //  65536 B
  _Float16* q16    = (_Float16*)(w + 98304);     // 4 MiB
  _Float16* k16    = (_Float16*)(w + 4292608);   // 4 MiB
  _Float16* v16p   = (_Float16*)(w + 8486912);   // 4 MiB
  float*    aop    = (float*)   (w + 12681216);  // 8 MiB

  k_prep<<<dim3(128), dim3(256), 0, stream>>>(in_w, out_w, Wg, Mg,
                                              w_in16, ow16, wh16);
  k_qkv<<<dim3(512), dim3(128), 0, stream>>>(inputs, in_b, sigma, w_in16,
                                             q16, k16, v16p);
  k_attn<<<dim3(32, 16), dim3(128), 0, stream>>>(q16, k16, v16p, aop);
  k_euler<<<dim3(512), dim3(128), 0, stream>>>(aop, out_b, ow16, wh16,
                                               bg, sigma, out);
}
